// SequentialChart_87737591922867
// MI455X (gfx1250) — matrix operand order repaired and checked by reading, not compiled, not run on hardware
//
#include <hip/hip_runtime.h>
#include <hip/hip_bf16.h>
#include <stdint.h>

// ---------------------------------------------------------------------------
// TreeLSTM chart fill for MI455X (gfx1250, wave32, WMMA).
//
// Insight: ops indices < start_index, writes at >= start_index  =>  all
// "sequential" steps are independent. Flatten to 131072 rows of
// gather -> (1x512)@(512x1280 Uᵀ) -> gates -> write.
//
// GEMM runs on v_wmma_f32_16x16x32_f16 (f16 in, f32 accumulate): 172 GFLOP
// lands near the ~54us HBM roofline instead of being fp32-matrix bound.
// U-block staging (L2 -> LDS) is double-buffered with CDNA5 async-to-LDS
// loads (ASYNCcnt) so it overlaps the WMMA stream.
// ---------------------------------------------------------------------------

typedef _Float16 half_t;
typedef __attribute__((ext_vector_type(16))) _Float16 v16h;
typedef __attribute__((ext_vector_type(8)))  _Float16 v8h;
typedef __attribute__((ext_vector_type(4)))  _Float16 v4h;
typedef __attribute__((ext_vector_type(2)))  _Float16 v2h;
typedef __attribute__((ext_vector_type(8)))  float    v8f;

#define HDIM      256       // H
#define TWO_H     512
#define BS        256
#define CHART_LEN 1024
#define NUM_STEPS 512
#define START_IDX 512
#define KDIM      512       // hh width (GEMM K)
#define MROWS     128       // output rows per workgroup (8 M-tiles of 16)
#define HH_STRIDE 520       // 512 + 8 halves pad (bank spread)
#define UB_ROWS   80        // 5 gates * 16 cols
#define UB_STRIDE 520
#define NJT       (HDIM / 16)   // 16 column tiles

#define HH_BYTES  (MROWS * HH_STRIDE * 2)                // 133120
#define UB_BYTES  (UB_ROWS * UB_STRIDE * 2)              // 83200
#define OPS_BYTES (MROWS * 2 * 4)                        // 1024
#define MAIN_SMEM (HH_BYTES + 2 * UB_BYTES + OPS_BYTES)  // 300544 (< 320KB)

static __device__ __forceinline__ v16h cat8(v8h a, v8h b) {
    return __builtin_shufflevector(a, b, 0, 1, 2, 3, 4, 5, 6, 7,
                                         8, 9, 10, 11, 12, 13, 14, 15);
}

static __device__ __forceinline__ float sigf(float x) {
    return 1.0f / (1.0f + __expf(-x));
}

// Issue async L2->LDS copy of the U block for column tile jt into buf.
// 80 rows x 512 halves, 16B (b128) per lane per op, 20 ops per thread.
// Tracked by ASYNCcnt; consumer does s_wait_asynccnt + barrier.
static __device__ __forceinline__ void issue_ublock(const half_t* __restrict__ U16,
                                                    half_t* buf, int jt, int tid) {
    for (int c = tid; c < (UB_ROWS * KDIM) / 8; c += 256) {   // 5120 chunks
        int br = c >> 6;               // block row 0..79 (64 chunks per row)
        int kc = (c & 63) << 3;        // 8-half (16B) chunk
        int g  = br >> 4;
        int nn = br & 15;
        uint64_t gaddr =
            (uint64_t)(uintptr_t)(U16 + (size_t)(g * HDIM + jt * 16 + nn) * KDIM + kc);
        uint32_t ldsoff = (uint32_t)(uintptr_t)(buf + br * UB_STRIDE + kc);
        asm volatile("global_load_async_to_lds_b128 %0, %1, off"
                     :: "v"(ldsoff), "v"(gaddr) : "memory");
    }
}

// Cast U (5H x 2H fp32) to f16 in workspace, 4 elements per thread.
__global__ void cast_u_kernel(const float* __restrict__ U,
                              half_t* __restrict__ U16, int n4) {
    int i = blockIdx.x * blockDim.x + threadIdx.x;
    if (i < n4) {
        float4 v = ((const float4*)U)[i];
        v4h h = { (half_t)v.x, (half_t)v.y, (half_t)v.z, (half_t)v.w };
        ((v4h*)U16)[i] = h;
    }
}

__global__ __launch_bounds__(256)
void chart_wmma_kernel(const float* __restrict__ chart,
                       const int*   __restrict__ ops,      // (steps, bs, 2)
                       const half_t* __restrict__ U16,     // (1280, 512) f16
                       const float* __restrict__ bias,     // (1280,)
                       float* __restrict__ out) {
    extern __shared__ char smem[];
    half_t* sHH  = (half_t*)smem;                           // [128][520] f16
    half_t* sUB0 = (half_t*)(smem + HH_BYTES);              // [80][520] f16
    half_t* sUB1 = (half_t*)(smem + HH_BYTES + UB_BYTES);   // [80][520] f16
    int*    sOpL = (int*)(smem + HH_BYTES + 2 * UB_BYTES);  // [128]
    int*    sOpR = sOpL + MROWS;

    const int tid  = threadIdx.x;
    const int row0 = blockIdx.x * MROWS;   // flat row = step*256 + batch

    // ---- phase 0: op indices for the 128 rows this WG owns -------------
    if (tid < MROWS) {
        int row  = row0 + tid;
        int step = row >> 8;
        int b    = row & (BS - 1);
        size_t o = ((size_t)step * BS + b) * 2;
        sOpL[tid] = ops[o];
        sOpR[tid] = ops[o + 1];
    }
    __syncthreads();

    // ---- phase 1: gather h-halves of left/right rows -> f16 LDS --------
    // hh[m][0:256] = left[H:2H],  hh[m][256:512] = right[H:2H]
    // 128 rows * 512 cols, 2 consecutive cols per thread (coalesced b64).
    for (int it = 0; it < (MROWS * KDIM) / (2 * 256); ++it) {
        int e  = (tid + it * 256) * 2;       // even col index
        int m  = e >> 9;                     // local row 0..127
        int k  = e & (KDIM - 1);
        int b  = (row0 + m) & (BS - 1);
        int op = (k < HDIM) ? sOpL[m] : sOpR[m];
        int kk = k & (HDIM - 1);
        const float* src =
            chart + (((size_t)b * CHART_LEN + op) * TWO_H + HDIM + kk);
        float2 v = *(const float2*)src;
        v2h h = { (half_t)v.x, (half_t)v.y };
        *(v2h*)(&sHH[m * HH_STRIDE + k]) = h;
    }
    __syncthreads();

    // Prime the async pipeline with the first U block.
    issue_ublock(U16, sUB0, 0, tid);

    const int wave = tid >> 5;   // 8 waves: wave w owns M-tile w
    const int lane = tid & 31;
    const int lrow = lane & 15;  // N / M sub-index within fragment
    const int lhi  = lane >> 4;  // half-wave select

    for (int jt = 0; jt < NJT; ++jt) {   // 16 column tiles of H
        // Wait for this wave's async copies (buffer jt&1), then make the
        // buffer visible WG-wide. Every wave waits its own ASYNCcnt before
        // the barrier, so post-barrier the whole block is resident.
        asm volatile("s_wait_asynccnt 0x0" ::: "memory");
        __syncthreads();

        // Overlap: kick off the next tile's L2->LDS copy behind the WMMAs.
        // (Safe: buffer (jt+1)&1 was last *read* in iteration jt-1, and all
        //  waves have passed that point thanks to the barrier above.)
        if (jt + 1 < NJT)
            issue_ublock(U16, (jt & 1) ? sUB0 : sUB1, jt + 1, tid);

        const half_t* ub = (jt & 1) ? sUB1 : sUB0;

        // ---- 5 gate accumulators, one 16x16 tile, K = 512 -------------
        v8f acc[5] = {};
        const half_t* hhbase = sHH + (wave * 16 + lrow) * HH_STRIDE;
        for (int kk = 0; kk < KDIM / 32; ++kk) {
            // A frag (16x32 f16): m = lane%16, 16 contiguous K per lane
            int ka = kk * 32 + lhi * 16;
            v8h alo = *(const v8h*)(hhbase + ka);
            v8h ahi = *(const v8h*)(hhbase + ka + 8);
            v16h afrag = cat8(alo, ahi);
            // B frag (32x16 f16): n = lane%16, 16 contiguous K per lane
            int kb = kk * 32 + lhi * 16;
#pragma unroll
            for (int g = 0; g < 5; ++g) {
                const half_t* bp = ub + (g * 16 + lrow) * UB_STRIDE + kb;
                v16h bfrag = cat8(*(const v8h*)bp, *(const v8h*)(bp + 8));
                acc[g] = __builtin_amdgcn_wmma_f32_16x16x32_f16(
                    false, afrag, false, bfrag, (short)0, acc[g],
                    false, false);
            }
        }

        // ---- epilogue: gates fused on accumulator registers -----------
        // C/D layout: vgpr r, lane -> M = r + 8*(lane>=16), N = lane%16
        int col = jt * 16 + lrow;
        float b_i  = bias[col];
        float b_fL = bias[HDIM + col];
        float b_fR = bias[2 * HDIM + col];
        float b_o  = bias[3 * HDIM + col];
        float b_u  = bias[4 * HDIM + col];
#pragma unroll
        for (int r = 0; r < 8; ++r) {
            int m    = wave * 16 + r + lhi * 8;
            int grow = row0 + m;
            int step = grow >> 8;
            int bidx = grow & (BS - 1);
            size_t cbase = (size_t)bidx * CHART_LEN;
            float ccL = chart[(cbase + sOpL[m]) * TWO_H + col];
            float ccR = chart[(cbase + sOpR[m]) * TWO_H + col];
            float gi  = sigf(acc[0][r] + b_i);
            float gfL = sigf(acc[1][r] + b_fL);
            float gfR = sigf(acc[2][r] + b_fR);
            float go  = sigf(acc[3][r] + b_o);
            float gu  = tanhf(acc[4][r] + b_u);
            float c   = gfL * ccL + gfR * ccR + gi * gu;
            float h   = go * tanhf(c);
            float* orow = out + (cbase + (size_t)(START_IDX + step)) * TWO_H;
            orow[col]        = c;
            orow[HDIM + col] = h;
        }
        // No trailing barrier: next iteration's wait+barrier fences reuse.
    }
}

extern "C" void kernel_launch(void* const* d_in, const int* in_sizes, int n_in,
                              void* d_out, int out_size, void* d_ws,
                              size_t ws_size, hipStream_t stream) {
    const float* chart = (const float*)d_in[0];
    const int*   ops   = (const int*)d_in[1];   // JAX default config: int32
    const float* U     = (const float*)d_in[3];
    const float* bias  = (const float*)d_in[4];
    float*  out = (float*)d_out;
    half_t* U16 = (half_t*)d_ws;                // 1280*512*2 = 1.25 MB
    (void)n_in; (void)out_size; (void)ws_size;

    // Output = input chart with rows [512,1024) overwritten: copy all,
    // compute kernel rewrites the second half. D2D async is capture-safe.
    size_t chart_bytes = (size_t)in_sizes[0] * sizeof(float);
    hipMemcpyAsync(out, chart, chart_bytes, hipMemcpyDeviceToDevice, stream);

    // U fp32 -> f16 workspace.
    int n4 = (5 * HDIM * TWO_H) / 4;
    cast_u_kernel<<<(n4 + 255) / 256, 256, 0, stream>>>(U, U16, n4);

    // Allow > default dynamic LDS (294 KB of the 320 KB WGP budget).
    hipFuncSetAttribute(reinterpret_cast<const void*>(chart_wmma_kernel),
                        hipFuncAttributeMaxDynamicSharedMemorySize, MAIN_SMEM);

    // 131072 rows / 128 per WG = 1024 workgroups of 8 waves.
    chart_wmma_kernel<<<(NUM_STEPS * BS) / MROWS, 256, MAIN_SMEM, stream>>>(
        chart, ops, U16, bias, out);
}
